// StyleBezierFusionModule_35399120454142
// MI455X (gfx1250) — matrix operand order repaired and checked by reading, not compiled, not run on hardware
//
#include <hip/hip_runtime.h>
#include <hip/hip_bf16.h>

typedef __attribute__((ext_vector_type(8)))  _Float16 v8h;
typedef __attribute__((ext_vector_type(16))) _Float16 v16h;
typedef __attribute__((ext_vector_type(8)))  float    v8f;

namespace {
constexpr int kB   = 4;
constexpr int kN   = 2048;
constexpr int kC   = 3072;
constexpr int kDS  = 256;
constexpr int kHID = 512;
constexpr int kM   = kB * kN;   // 8192 tokens
constexpr int kC2  = 2 * kC;    // 6144
constexpr float kEps = 1e-5f;
constexpr int kNCH = 8;         // N-chunks for channel-stat reduction

// ---- workspace layout (bytes) ----
constexpr size_t OFF_ATT  = 0;              // f32 att [8192x3072]; later reused as f (f32)
constexpr size_t OFF_W1T  = 100663296;      // f16 fn_w1^T [3072x6144]
constexpr size_t OFF_W2T  = 138412032;      // f16 fn_w2^T [3072x3072]
constexpr size_t OFF_W3T  = 157286400;      // f16 op_w1^T
constexpr size_t OFF_W4T  = 176160768;      // f16 op_w2^T
constexpr size_t OFF_AF   = 195035136;      // f16 fused_in [8192x6144]; later FLN + G1
constexpr size_t OFF_H1   = 295698432;      // f16 h1 [8192x3072]
constexpr size_t OFF_S    = 346030080;      // f32 s [4x256]
constexpr size_t OFF_OVEC = 346034176;      // f32 o_vec [3x4x3072]
constexpr size_t OFF_GH   = 346181632;      // f32 gate hidden [4x3072]
constexpr size_t OFF_GATE = 346230784;      // f32 gate [4x3072]
constexpr size_t OFF_ALPHA= 346279936;      // f32 alpha [4x3072]
constexpr size_t OFF_BETA = 346329088;      // f32 beta  [4x3072]
constexpr size_t OFF_PS   = 346378240;      // f32 partial sum   [4x8x3072]
constexpr size_t OFF_PS2  = 346771456;      // f32 partial sumsq [4x8x3072]
constexpr size_t OFF_FLN  = OFF_AF;                 // f16 LN(f) [8192x3072]
constexpr size_t OFF_G1   = OFF_AF + 50331648;      // f16 g1    [8192x3072]
} // namespace

// ---------------- CDNA5 async global->LDS helpers ----------------
__device__ __forceinline__ void async_b128(uint32_t lds_addr, const void* gaddr) {
  // GLOBAL_LOAD_ASYNC_TO_LDS_B128: per-lane 16B DMA global -> LDS, tracked by ASYNCcnt
  asm volatile("global_load_async_to_lds_b128 %0, %1, off"
               :: "v"(lds_addr), "v"(gaddr) : "memory");
}
__device__ __forceinline__ void wait_async_le4() {
  asm volatile("s_wait_asynccnt 0x4" ::: "memory");
}
__device__ __forceinline__ void wait_async_le0() {
  asm volatile("s_wait_asynccnt 0x0" ::: "memory");
}
__device__ __forceinline__ uint32_t lds_addr32(const void* p) {
  // flat LDS addresses: low 32 bits are the LDS byte offset (ISA flat aperture rule)
  return (uint32_t)(unsigned long long)p;
}

// ---------------- reduction helper (wave32-aware) ----------------
__device__ __forceinline__ void blockReduce2(float& a, float& b, float* sm) {
  #pragma unroll
  for (int o = 16; o > 0; o >>= 1) {
    a += __shfl_down(a, o, 32);
    b += __shfl_down(b, o, 32);
  }
  const int lane = threadIdx.x & 31;
  const int w    = threadIdx.x >> 5;
  const int nw   = (blockDim.x + 31) >> 5;
  if (lane == 0) { sm[w] = a; sm[32 + w] = b; }
  __syncthreads();
  if (w == 0) {
    a = (lane < nw) ? sm[lane]      : 0.f;
    b = (lane < nw) ? sm[32 + lane] : 0.f;
    #pragma unroll
    for (int o = 16; o > 0; o >>= 1) {
      a += __shfl_down(a, o, 32);
      b += __shfl_down(b, o, 32);
    }
    if (lane == 0) { sm[0] = a; sm[32] = b; }
  }
  __syncthreads();
  a = sm[0]; b = sm[32];
  __syncthreads();
}

// ---------------- style encoder: s = LN(LN(MLP(style))) ----------------
__global__ __launch_bounds__(512)
void style_enc_kernel(const float* __restrict__ style,
                      const float* __restrict__ w1, const float* __restrict__ b1,
                      const float* __restrict__ w2, const float* __restrict__ b2,
                      const float* __restrict__ w3, const float* __restrict__ b3,
                      const float* __restrict__ lg1, const float* __restrict__ lb1,
                      const float* __restrict__ lg2, const float* __restrict__ lb2,
                      float* __restrict__ s_out) {
  __shared__ float sbuf[kDS];
  __shared__ float hbuf[kHID];
  __shared__ float red[64];
  const int b = blockIdx.x, tid = threadIdx.x;
  if (tid < kDS) sbuf[tid] = style[b * kDS + tid];
  __syncthreads();
  float a = b1[tid];
  for (int k = 0; k < kDS; ++k) a += sbuf[k] * w1[k * kHID + tid];
  hbuf[tid] = fmaxf(a, 0.f);
  __syncthreads();
  a = b2[tid];
  for (int k = 0; k < kHID; ++k) a += hbuf[k] * w2[k * kHID + tid];
  __syncthreads();
  hbuf[tid] = fmaxf(a, 0.f);
  __syncthreads();
  if (tid < kDS) {
    a = b3[tid];
    for (int k = 0; k < kHID; ++k) a += hbuf[k] * w3[k * kDS + tid];
  }
  __syncthreads();
  if (tid < kDS) sbuf[tid] = a;
  __syncthreads();
  float x = (tid < kDS) ? sbuf[tid] : 0.f;
  float s1 = x, s2 = x * x;
  blockReduce2(s1, s2, red);
  float mean = s1 / kDS, var = s2 / kDS - mean * mean;
  float rs = rsqrtf(var + kEps);
  if (tid < kDS) sbuf[tid] = (x - mean) * rs * lg1[tid] + lb1[tid];
  __syncthreads();
  x = (tid < kDS) ? sbuf[tid] : 0.f;
  s1 = x; s2 = x * x;
  blockReduce2(s1, s2, red);
  mean = s1 / kDS; var = s2 / kDS - mean * mean;
  rs = rsqrtf(var + kEps);
  if (tid < kDS) s_out[b * kDS + tid] = (x - mean) * rs * lg2[tid] + lb2[tid];
}

// ------- per-batch attention output vector: o_vec[i,b,:] = (s@wv_i)@wo_i + bo_i -------
__global__ __launch_bounds__(512)
void vo_kernel(const float* __restrict__ s, const int* __restrict__ mask,
               const float* __restrict__ wv0, const float* __restrict__ wv1, const float* __restrict__ wv2,
               const float* __restrict__ wo0, const float* __restrict__ wo1, const float* __restrict__ wo2,
               const float* __restrict__ bo0, const float* __restrict__ bo1, const float* __restrict__ bo2,
               float* __restrict__ ovec) {
  __shared__ float sv[kDS];
  __shared__ float vv[kHID];
  const int i = blockIdx.x, b = blockIdx.y, tid = threadIdx.x;
  const float* wv = (i == 0) ? wv0 : (i == 1) ? wv1 : wv2;
  const float* wo = (i == 0) ? wo0 : (i == 1) ? wo1 : wo2;
  const float* bo = (i == 0) ? bo0 : (i == 1) ? bo1 : bo2;
  if (tid < kDS) sv[tid] = s[b * kDS + tid];
  __syncthreads();
  float a = 0.f;  // no bias on v in reference
  for (int k = 0; k < kDS; ++k) a += sv[k] * wv[k * kHID + tid];
  vv[tid] = a;
  __syncthreads();
  // softmax over the single style token == 1 exactly -> av == v; mask==0 -> NaN
  const float nanmul = (mask[b] == 0) ? __uint_as_float(0x7fc00000u) : 1.0f;
  for (int c = tid; c < kC; c += kHID) {
    float o = bo[c];
    for (int j = 0; j < kHID; ++j) o += vv[j] * wo[(size_t)j * kC + c];
    ovec[((size_t)i * kB + b) * kC + c] = o * nanmul;
  }
}

// ------- gate hidden: gh = relu(s @ sm_w1 + sm_b1) -------
__global__ __launch_bounds__(256)
void gateh_kernel(const float* __restrict__ s, const float* __restrict__ w,
                  const float* __restrict__ bias, float* __restrict__ gh) {
  __shared__ float sv[kDS];
  const int b = blockIdx.y, c = blockIdx.x * 256 + threadIdx.x;
  if (threadIdx.x < kDS) sv[threadIdx.x] = s[b * kDS + threadIdx.x];
  __syncthreads();
  float a = bias[c];
  for (int k = 0; k < kDS; ++k) a += sv[k] * w[(size_t)k * kC + c];
  gh[(size_t)b * kC + c] = fmaxf(a, 0.f);
}

// ------- gate = sigmoid(gh @ sm_w2 + sm_b2) -------
__global__ __launch_bounds__(256)
void gate_kernel(const float* __restrict__ gh, const float* __restrict__ w,
                 const float* __restrict__ bias, float* __restrict__ gate) {
  __shared__ float hv[kC];
  const int b = blockIdx.y, c = blockIdx.x * 256 + threadIdx.x;
  for (int j = 0; j < kC / 256; ++j) hv[threadIdx.x + j * 256] = gh[(size_t)b * kC + threadIdx.x + j * 256];
  __syncthreads();
  float a = bias[c];
  for (int k = 0; k < kC; ++k) a += hv[k] * w[(size_t)k * kC + c];
  gate[(size_t)b * kC + c] = 1.0f / (1.0f + expf(-a));
}

// ------- att = sum_i mix_i * LN_i(LN(cf) + o_vec_i) -------
__global__ __launch_bounds__(256)
void lnatt_kernel(const float* __restrict__ cf, const float* __restrict__ ovec,
                  const float* __restrict__ mix,
                  const float* __restrict__ cng, const float* __restrict__ cnb,
                  const float* __restrict__ g0, const float* __restrict__ b0,
                  const float* __restrict__ g1, const float* __restrict__ b1,
                  const float* __restrict__ g2, const float* __restrict__ b2,
                  float* __restrict__ att) {
  constexpr int PER = kC / 256;  // 12
  __shared__ float red[64];
  const int token = blockIdx.x;
  const int b = token / kN;
  const int tid = threadIdx.x;
  const float* row = cf + (size_t)token * kC;
  float xv[PER];
  float s1 = 0.f, s2 = 0.f;
  #pragma unroll
  for (int j = 0; j < PER; ++j) {
    const float v = row[tid + j * 256];
    xv[j] = v; s1 += v; s2 += v * v;
  }
  blockReduce2(s1, s2, red);
  float mean = s1 / kC, var = s2 / kC - mean * mean;
  float rs = rsqrtf(var + kEps);
  #pragma unroll
  for (int j = 0; j < PER; ++j) {
    const int c = tid + j * 256;
    xv[j] = (xv[j] - mean) * rs * cng[c] + cnb[c];
  }
  const float m0 = mix[0], m1 = mix[1], m2 = mix[2];
  const float mmax = fmaxf(m0, fmaxf(m1, m2));
  const float e0 = expf(m0 - mmax), e1 = expf(m1 - mmax), e2 = expf(m2 - mmax);
  const float esum = e0 + e1 + e2;
  const float mw[3] = {e0 / esum, e1 / esum, e2 / esum};
  const float* const lg[3] = {g0, g1, g2};
  const float* const lb[3] = {b0, b1, b2};
  float accv[PER];
  #pragma unroll
  for (int j = 0; j < PER; ++j) accv[j] = 0.f;
  for (int i = 0; i < 3; ++i) {
    const float* ov = ovec + ((size_t)i * kB + b) * kC;
    float yv[PER];
    s1 = 0.f; s2 = 0.f;
    #pragma unroll
    for (int j = 0; j < PER; ++j) {
      const int c = tid + j * 256;
      const float y = xv[j] + ov[c];
      yv[j] = y; s1 += y; s2 += y * y;
    }
    blockReduce2(s1, s2, red);
    mean = s1 / kC; var = s2 / kC - mean * mean;
    rs = rsqrtf(var + kEps);
    #pragma unroll
    for (int j = 0; j < PER; ++j) {
      const int c = tid + j * 256;
      accv[j] += mw[i] * ((yv[j] - mean) * rs * lg[i][c] + lb[i][c]);
    }
  }
  float* orow = att + (size_t)token * kC;
  #pragma unroll
  for (int j = 0; j < PER; ++j) orow[tid + j * 256] = accv[j];
}

// ------- channel stats over N, phase 1: partial sums over 256-row chunks -------
__global__ __launch_bounds__(256)
void chanstats_part_kernel(const float* __restrict__ att,
                           float* __restrict__ psum, float* __restrict__ psumsq) {
  const int b = blockIdx.z, nc = blockIdx.y;
  const int c = blockIdx.x * 256 + threadIdx.x;
  constexpr int nrows = kN / kNCH;  // 256
  const float* base = att + ((size_t)b * kN + (size_t)nc * nrows) * kC + c;
  float s1 = 0.f, s2 = 0.f;
  for (int n = 0; n < nrows; ++n) {
    const float v = base[(size_t)n * kC];
    s1 += v; s2 += v * v;
  }
  const size_t idx = ((size_t)b * kNCH + nc) * kC + c;
  psum[idx] = s1;
  psumsq[idx] = s2;
}

// ------- channel stats phase 2: combine -> stylized affine (att*alpha + beta) -------
__global__ __launch_bounds__(256)
void chanstats_fin_kernel(const float* __restrict__ psum, const float* __restrict__ psumsq,
                          const float* __restrict__ gate,
                          float* __restrict__ alpha, float* __restrict__ beta) {
  const int b = blockIdx.y, c = blockIdx.x * 256 + threadIdx.x;
  float s1 = 0.f, s2 = 0.f;
  #pragma unroll
  for (int nc = 0; nc < kNCH; ++nc) {
    const size_t idx = ((size_t)b * kNCH + nc) * kC + c;
    s1 += psum[idx]; s2 += psumsq[idx];
  }
  const float mean = s1 / kN;
  const float var = (s2 - kN * mean * mean) / (kN - 1);  // ddof=1
  const float g = gate[(size_t)b * kC + c];
  const float cs = sqrtf(var + kEps);
  const float ss = sqrtf(g * g * var + kEps);
  const float al = ss / cs;
  alpha[(size_t)b * kC + c] = al;
  beta [(size_t)b * kC + c] = mean * g - mean * al;
}

// ------- fused_in (f16): [att | att*alpha+beta] -------
__global__ __launch_bounds__(256)
void prep_fused_kernel(const float* __restrict__ att, const float* __restrict__ alpha,
                       const float* __restrict__ beta, _Float16* __restrict__ af) {
  const int m = blockIdx.y, c = blockIdx.x * 256 + threadIdx.x;
  const int b = m >> 11;  // m / 2048
  const float a = att[(size_t)m * kC + c];
  af[(size_t)m * kC2 + c]      = (_Float16)a;
  af[(size_t)m * kC2 + kC + c] = (_Float16)(a * alpha[(size_t)b * kC + c] + beta[(size_t)b * kC + c]);
}

// ------- weight convert + transpose: W[K][N] f32 -> Wt[N][K] f16 -------
__global__ __launch_bounds__(256)
void wconv_kernel(const float* __restrict__ W, _Float16* __restrict__ Wt, int K, int N) {
  __shared__ float tile[32][33];
  const int k0 = blockIdx.y * 32, n0 = blockIdx.x * 32;
  const int tx = threadIdx.x, ty = threadIdx.y;
  for (int r = ty; r < 32; r += 8) tile[r][tx] = W[(size_t)(k0 + r) * N + n0 + tx];
  __syncthreads();
  for (int r = ty; r < 32; r += 8) Wt[(size_t)(n0 + r) * K + k0 + tx] = (_Float16)tile[tx][r];
}

// ------- LN over f rows -> f16 -------
__global__ __launch_bounds__(256)
void lnf_kernel(const float* __restrict__ f, const float* __restrict__ g,
                const float* __restrict__ bb, _Float16* __restrict__ out) {
  constexpr int PER = kC / 256;
  __shared__ float red[64];
  const int token = blockIdx.x, tid = threadIdx.x;
  const float* row = f + (size_t)token * kC;
  float xv[PER];
  float s1 = 0.f, s2 = 0.f;
  #pragma unroll
  for (int j = 0; j < PER; ++j) {
    const float v = row[tid + j * 256];
    xv[j] = v; s1 += v; s2 += v * v;
  }
  blockReduce2(s1, s2, red);
  const float mean = s1 / kC, var = s2 / kC - mean * mean;
  const float rs = rsqrtf(var + kEps);
  _Float16* orow = out + (size_t)token * kC;
  #pragma unroll
  for (int j = 0; j < PER; ++j) {
    const int c = tid + j * 256;
    orow[c] = (_Float16)((xv[j] - mean) * rs * g[c] + bb[c]);
  }
}

// ---------------- WMMA GEMM with async double-buffered LDS pipeline ----------------
// A: M x K f16 row-major (activations).  Bt: N x K f16 row-major (pre-transposed weight).
// 256 threads = 8 waves; block tile 128x128, stage depth BKT=64 (2 WMMA K-chunks);
// wave tile 32x64 (2x4 WMMA f32 accumulators).
#define BMT 128
#define BNT 128
#define BKT 64

union HalfVec { v16h v; struct { v8h lo, hi; } p; };

template <bool RELU, bool F16OUT, bool RES>
__global__ __launch_bounds__(256)
void gemm_kernel(const _Float16* __restrict__ A, const _Float16* __restrict__ Bt,
                 const float* __restrict__ bias, const float* __restrict__ res,
                 void* __restrict__ outv, int M, int N, int K) {
  __shared__ __align__(16) _Float16 sA[2][BMT * BKT];  // 2 x 16 KB
  __shared__ __align__(16) _Float16 sB[2][BNT * BKT];  // 2 x 16 KB
  const int tid  = threadIdx.x;
  const int lane = tid & 31;
  const int wave = tid >> 5;
  const int m0 = blockIdx.y * BMT;
  const int n0 = blockIdx.x * BNT;
  const int wm = (wave & 3) * 32;   // 0,32,64,96
  const int wn = (wave >> 2) * 64;  // 0,64
  const int am = lane & 15;
  const int ah = lane >> 4;         // K-half select per WMMA operand layout
  v8f acc[2][4] = {};

  // stage loader: tile = 128 rows x 64 halfs = 512 x 16-half chunks;
  // thread t moves chunks {t, t+256} of A and of B (4 async b128 per stage)
  const int r0 = tid >> 2;            // rows 0..63 (second chunk: +64)
  const int c0 = (tid & 3) * 16;      // half offset within row
  const uint32_t ldsA = lds_addr32(&sA[0][0]);
  const uint32_t ldsB = lds_addr32(&sB[0][0]);
  constexpr uint32_t stageB = BMT * BKT * 2;  // 16384 bytes per stage buffer

  const _Float16* gA0 = A  + (size_t)(m0 + r0)      * K + c0;
  const _Float16* gA1 = A  + (size_t)(m0 + r0 + 64) * K + c0;
  const _Float16* gB0 = Bt + (size_t)(n0 + r0)      * K + c0;
  const _Float16* gB1 = Bt + (size_t)(n0 + r0 + 64) * K + c0;
  const uint32_t lA0 = ldsA + (uint32_t)(r0 * BKT + c0) * 2u;
  const uint32_t lA1 = lA0 + 64u * BKT * 2u;
  const uint32_t lB0 = ldsB + (uint32_t)(r0 * BKT + c0) * 2u;
  const uint32_t lB1 = lB0 + 64u * BKT * 2u;

  const int T = K / BKT;
  // prologue: stage 0 -> buffer 0
  async_b128(lA0, gA0);
  async_b128(lA1, gA1);
  async_b128(lB0, gB0);
  async_b128(lB1, gB1);

  for (int kt = 0; kt < T; ++kt) {
    const int p = kt & 1;
    if (kt + 1 < T) {
      const size_t ko = (size_t)(kt + 1) * BKT;
      const uint32_t po = (uint32_t)(p ^ 1) * stageB;
      async_b128(lA0 + po, gA0 + ko);
      async_b128(lA1 + po, gA1 + ko);
      async_b128(lB0 + po, gB0 + ko);
      async_b128(lB1 + po, gB1 + ko);
      wait_async_le4();   // async loads retire in order -> stage kt resident
    } else {
      wait_async_le0();
    }
    __syncthreads();      // all waves' stage-kt data visible

    #pragma unroll
    for (int kk = 0; kk < 2; ++kk) {  // two 16x16x32 K-chunks per stage
      v16h aF[2];
      v16h bF[4];
      #pragma unroll
      for (int tm = 0; tm < 2; ++tm) {
        const _Float16* pa = &sA[p][(wm + tm * 16 + am) * BKT + kk * 32 + ah * 8];
        HalfVec u;
        u.p.lo = *(const v8h*)(pa);
        u.p.hi = *(const v8h*)(pa + 16);
        aF[tm] = u.v;
      }
      #pragma unroll
      for (int tn = 0; tn < 4; ++tn) {
        const _Float16* pb = &sB[p][(wn + tn * 16 + am) * BKT + kk * 32 + ah * 8];
        HalfVec u;
        u.p.lo = *(const v8h*)(pb);
        u.p.hi = *(const v8h*)(pb + 16);
        bF[tn] = u.v;
      }
      #pragma unroll
      for (int tn = 0; tn < 4; ++tn) {
        #pragma unroll
        for (int tm = 0; tm < 2; ++tm) {
          acc[tm][tn] = __builtin_amdgcn_wmma_f32_16x16x32_f16(
              false, aF[tm], false, bF[tn], (short)0, acc[tm][tn], false, false);
        }
      }
    }
    __syncthreads();      // stage-kt buffer free for refill at stage kt+2
  }

  // epilogue; C layout: VGPR e, lanes 0-15 -> M=e, lanes 16-31 -> M=e+8; N=lane%16
  float* outF = (float*)outv;
  _Float16* outH = (_Float16*)outv;
  const int rowBase = m0 + wm + ((lane >> 4) << 3);
  const int cb = lane & 15;
  #pragma unroll
  for (int tm = 0; tm < 2; ++tm) {
    #pragma unroll
    for (int tn = 0; tn < 4; ++tn) {
      const int col = n0 + wn + tn * 16 + cb;
      const float bv = bias[col];
      #pragma unroll
      for (int e = 0; e < 8; ++e) {
        const int row = rowBase + tm * 16 + e;
        float v = acc[tm][tn][e] + bv;
        if (RELU) v = fmaxf(v, 0.f);
        if (RES)  v += res[(size_t)row * N + col];
        if (F16OUT) outH[(size_t)row * N + col] = (_Float16)v;
        else        outF[(size_t)row * N + col] = v;
      }
    }
  }
}

extern "C" void kernel_launch(void* const* d_in, const int* in_sizes, int n_in,
                              void* d_out, int out_size, void* d_ws, size_t ws_size,
                              hipStream_t stream) {
  (void)in_sizes; (void)n_in; (void)out_size; (void)ws_size;
  const float* cf    = (const float*)d_in[0];
  const float* style = (const float*)d_in[1];
  const int*   mask  = (const int*)  d_in[2];
  const float* se_w1 = (const float*)d_in[3];
  const float* se_b1 = (const float*)d_in[4];
  const float* se_w2 = (const float*)d_in[5];
  const float* se_b2 = (const float*)d_in[6];
  const float* se_w3 = (const float*)d_in[7];
  const float* se_b3 = (const float*)d_in[8];
  const float* se_lng= (const float*)d_in[9];
  const float* se_lnb= (const float*)d_in[10];
  const float* sn_g  = (const float*)d_in[11];
  const float* sn_b  = (const float*)d_in[12];
  // cma blocks: base 13 + 7*i : wq,wk,wv,wo,bo,ln_g,ln_b  (wq,wk unused: softmax over 1 key == 1)
  const float* wv0 = (const float*)d_in[15];
  const float* wo0 = (const float*)d_in[16];
  const float* bo0 = (const float*)d_in[17];
  const float* cg0 = (const float*)d_in[18];
  const float* cb0 = (const float*)d_in[19];
  const float* wv1 = (const float*)d_in[22];
  const float* wo1 = (const float*)d_in[23];
  const float* bo1 = (const float*)d_in[24];
  const float* cg1 = (const float*)d_in[25];
  const float* cb1 = (const float*)d_in[26];
  const float* wv2 = (const float*)d_in[29];
  const float* wo2 = (const float*)d_in[30];
  const float* bo2 = (const float*)d_in[31];
  const float* cg2 = (const float*)d_in[32];
  const float* cb2 = (const float*)d_in[33];
  const float* sm_w1 = (const float*)d_in[34];
  const float* sm_b1 = (const float*)d_in[35];
  const float* sm_w2 = (const float*)d_in[36];
  const float* sm_b2 = (const float*)d_in[37];
  const float* fn_w1 = (const float*)d_in[38];
  const float* fn_b1 = (const float*)d_in[39];
  const float* fn_w2 = (const float*)d_in[40];
  const float* fn_b2 = (const float*)d_in[41];
  const float* fn_lng= (const float*)d_in[42];
  const float* fn_lnb= (const float*)d_in[43];
  const float* op_w1 = (const float*)d_in[44];
  const float* op_b1 = (const float*)d_in[45];
  const float* op_w2 = (const float*)d_in[46];
  const float* op_b2 = (const float*)d_in[47];
  const float* cn_g  = (const float*)d_in[48];
  const float* cn_b  = (const float*)d_in[49];
  const float* mix   = (const float*)d_in[50];

  char* ws = (char*)d_ws;
  float*     att   = (float*)    (ws + OFF_ATT);
  _Float16*  w1t   = (_Float16*) (ws + OFF_W1T);
  _Float16*  w2t   = (_Float16*) (ws + OFF_W2T);
  _Float16*  w3t   = (_Float16*) (ws + OFF_W3T);
  _Float16*  w4t   = (_Float16*) (ws + OFF_W4T);
  _Float16*  af    = (_Float16*) (ws + OFF_AF);
  _Float16*  h1    = (_Float16*) (ws + OFF_H1);
  float*     sfin  = (float*)    (ws + OFF_S);
  float*     ovec  = (float*)    (ws + OFF_OVEC);
  float*     gh    = (float*)    (ws + OFF_GH);
  float*     gate  = (float*)    (ws + OFF_GATE);
  float*     alpha = (float*)    (ws + OFF_ALPHA);
  float*     beta  = (float*)    (ws + OFF_BETA);
  float*     psum  = (float*)    (ws + OFF_PS);
  float*     psumsq= (float*)    (ws + OFF_PS2);
  float*     fbuf  = (float*)    (ws + OFF_ATT);   // reuse att region for f
  _Float16*  fln   = (_Float16*) (ws + OFF_FLN);   // reuse af region
  _Float16*  g1    = (_Float16*) (ws + OFF_G1);    // reuse af region (+50MB)
  float*     out   = (float*)d_out;

  // --- weight conversion/transposition (f32 KxN -> f16 NxK) ---
  wconv_kernel<<<dim3(kC / 32, kC2 / 32), dim3(32, 8), 0, stream>>>(fn_w1, w1t, kC2, kC);
  wconv_kernel<<<dim3(kC / 32, kC  / 32), dim3(32, 8), 0, stream>>>(fn_w2, w2t, kC,  kC);
  wconv_kernel<<<dim3(kC / 32, kC  / 32), dim3(32, 8), 0, stream>>>(op_w1, w3t, kC,  kC);
  wconv_kernel<<<dim3(kC / 32, kC  / 32), dim3(32, 8), 0, stream>>>(op_w2, w4t, kC,  kC);

  // --- tiny style path ---
  style_enc_kernel<<<kB, 512, 0, stream>>>(style, se_w1, se_b1, se_w2, se_b2,
                                           se_w3, se_b3, se_lng, se_lnb, sn_g, sn_b, sfin);
  vo_kernel<<<dim3(3, kB), 512, 0, stream>>>(sfin, mask, wv0, wv1, wv2,
                                             wo0, wo1, wo2, bo0, bo1, bo2, ovec);
  gateh_kernel<<<dim3(kC / 256, kB), 256, 0, stream>>>(sfin, sm_w1, sm_b1, gh);
  gate_kernel<<<dim3(kC / 256, kB), 256, 0, stream>>>(gh, sm_w2, sm_b2, gate);

  // --- att (B,N,C) ---
  lnatt_kernel<<<kM, 256, 0, stream>>>(cf, ovec, mix, cn_g, cn_b,
                                       cg0, cb0, cg1, cb1, cg2, cb2, att);
  chanstats_part_kernel<<<dim3(kC / 256, kNCH, kB), 256, 0, stream>>>(att, psum, psumsq);
  chanstats_fin_kernel<<<dim3(kC / 256, kB), 256, 0, stream>>>(psum, psumsq, gate, alpha, beta);
  prep_fused_kernel<<<dim3(kC / 256, kM), 256, 0, stream>>>(att, alpha, beta, af);

  // --- GEMM chain (f16 WMMA, f32 accumulate, async LDS pipeline) ---
  const dim3 ggrid(kC / BNT, kM / BMT);  // (24, 64)
  gemm_kernel<true,  true,  false><<<ggrid, 256, 0, stream>>>(af,  w1t, fn_b1, nullptr, h1,   kM, kC, kC2);
  gemm_kernel<false, false, false><<<ggrid, 256, 0, stream>>>(h1,  w2t, fn_b2, nullptr, fbuf, kM, kC, kC);
  lnf_kernel<<<kM, 256, 0, stream>>>(fbuf, fn_lng, fn_lnb, fln);
  gemm_kernel<true,  true,  false><<<ggrid, 256, 0, stream>>>(fln, w3t, op_b1, nullptr, g1,   kM, kC, kC);
  gemm_kernel<false, false, true ><<<ggrid, 256, 0, stream>>>(g1,  w4t, op_b2, cf,      out,  kM, kC, kC);
}
